// TAttention_5497558139127
// MI455X (gfx1250) — matrix operand order repaired and checked by reading, not compiled, not run on hardware
//
#include <hip/hip_runtime.h>
#include <stdint.h>

// Problem constants (match reference)
#define B_  16
#define C_  64
#define N_  307
#define T_  288
#define R_  10

typedef __attribute__((ext_vector_type(16))) __bf16        v16bf;
typedef __attribute__((ext_vector_type(8)))  float          v8f;
typedef __attribute__((ext_vector_type(8)))  unsigned short ushort8;

union BfFrag {
    v16bf   bf;
    ushort8 h[2];
};

__device__ __forceinline__ unsigned short f2bf(float f) {
    union { float f; unsigned u; } v; v.f = f;
    unsigned u = v.u;
    // round-to-nearest-even fp32 -> bf16
    return (unsigned short)((u + 0x7FFFu + ((u >> 16) & 1u)) >> 16);
}

// ---------------------------------------------------------------------------
// Kernel 1: K[b][n][t] = sum_c alpha[c] * x[b][c][n][t]
// ---------------------------------------------------------------------------
__global__ __launch_bounds__(256)
void k_reduce_c(const float* __restrict__ x, const float* __restrict__ alpha,
                float* __restrict__ Kbuf) {
    int idx = blockIdx.x * blockDim.x + threadIdx.x;
    if (idx >= B_ * N_ * T_) return;
    int t = idx % T_;
    int n = (idx / T_) % N_;
    int b = idx / (N_ * T_);
    const float* xp = x + ((size_t)(b * C_) * N_ + n) * T_ + t;
    float acc = 0.f;
#pragma unroll 8
    for (int c = 0; c < C_; ++c)
        acc += alpha[c] * xp[(size_t)c * N_ * T_];
    Kbuf[idx] = acc;   // layout [b][n][t]
}

// ---------------------------------------------------------------------------
// Kernel 2: KW1[b][t][r] = sum_n K[b][n][t]*W1[r][n];  same for W2
// ---------------------------------------------------------------------------
__global__ __launch_bounds__(288)
void k_project(const float* __restrict__ Kbuf,
               const float* __restrict__ W1, const float* __restrict__ W2,
               float* __restrict__ KW1, float* __restrict__ QW2) {
    __shared__ float sW1[R_][N_];
    __shared__ float sW2[R_][N_];
    int b = blockIdx.x;
    for (int i = threadIdx.x; i < R_ * N_; i += blockDim.x) {
        sW1[i / N_][i % N_] = W1[i];
        sW2[i / N_][i % N_] = W2[i];
    }
    __syncthreads();
    int t = threadIdx.x;
    if (t >= T_) return;
    float a1[R_] = {};
    float a2[R_] = {};
    const float* kp = Kbuf + (size_t)b * N_ * T_ + t;
    for (int n = 0; n < N_; ++n) {
        float kv = kp[(size_t)n * T_];          // coalesced across t-lanes
#pragma unroll
        for (int r = 0; r < R_; ++r) {
            a1[r] += kv * sW1[r][n];
            a2[r] += kv * sW2[r][n];
        }
    }
#pragma unroll
    for (int r = 0; r < R_; ++r) {
        KW1[((size_t)b * T_ + t) * R_ + r] = a1[r];
        QW2[((size_t)b * T_ + t) * R_ + r] = a2[r];
    }
}

// ---------------------------------------------------------------------------
// Kernel 3: scores + softmax -> A_bf16[b][t][s]
// ---------------------------------------------------------------------------
__global__ __launch_bounds__(288)
void k_scores_softmax(const float* __restrict__ KW1, const float* __restrict__ QW2,
                      unsigned short* __restrict__ Abf) {
    int b = blockIdx.x / T_;
    int t = blockIdx.x % T_;
    __shared__ float kq[R_];
    __shared__ float red[512];
    if (threadIdx.x < R_) kq[threadIdx.x] = KW1[((size_t)b * T_ + t) * R_ + threadIdx.x];
    __syncthreads();

    int s = threadIdx.x;
    const float* q = QW2 + ((size_t)b * T_ + s) * R_;
    float score = 0.f;
#pragma unroll
    for (int r = 0; r < R_; ++r) score += kq[r] * q[r];

    // block max
    red[threadIdx.x] = score;
    for (int i = threadIdx.x + T_; i < 512; i += T_) red[i] = -1e30f;
    __syncthreads();
    for (int st = 256; st > 0; st >>= 1) {
        if (threadIdx.x < st) red[threadIdx.x] = fmaxf(red[threadIdx.x], red[threadIdx.x + st]);
        __syncthreads();
    }
    float m = red[0];
    __syncthreads();

    float e = __expf(score - m);
    red[threadIdx.x] = e;
    for (int i = threadIdx.x + T_; i < 512; i += T_) red[i] = 0.f;
    __syncthreads();
    for (int st = 256; st > 0; st >>= 1) {
        if (threadIdx.x < st) red[threadIdx.x] += red[threadIdx.x + st];
        __syncthreads();
    }
    float inv = 1.f / red[0];

    Abf[((size_t)b * T_ + t) * T_ + s] = f2bf(e * inv);   // coalesced in s
}

// ---------------------------------------------------------------------------
// Kernel 4: out[b][c][n][t] = sum_s x[b][c][n][s] * A[b][t][s]
// Block = one (b,c), tile BM=128 (n) x BN=288 (all of T), BK=32 (s).
// 8 waves (4x2); each wave: 2 M-frags x 9 N-frags of v_wmma_f32_16x16x32_bf16.
// x tile: fp32 global -> bf16 convert -> LDS (synchronous; needs VALU convert).
// A tile (B operand): double-buffered GLOBAL_LOAD_ASYNC_TO_LDS_B128 staged one
// K-step ahead (ASYNCcnt pipeline) -> removes the 4x redundant L2 reads.
// ---------------------------------------------------------------------------
#define BM 128
#define BK 32
#define XPITCH 40   // halfwords per LDS row of sX (32 + 8 pad -> 80B)
#define BPITCH 40   // halfwords per LDS row of sB

__global__ __launch_bounds__(256)
void k_av(const float* __restrict__ x,
          const unsigned short* __restrict__ Abf,
          float* __restrict__ out) {
    __shared__ __attribute__((aligned(16))) unsigned short sX[BM * XPITCH];
    __shared__ __attribute__((aligned(16))) unsigned short sB[2][T_ * BPITCH];

    const int m0 = blockIdx.x * BM;              // n-tile base (0,128,256)
    const int c  = blockIdx.y;
    const int b  = blockIdx.z;

    const float* xbc = x   + (size_t)(b * C_ + c) * N_ * T_;
    float*       obc = out + (size_t)(b * C_ + c) * N_ * T_;
    const unsigned short* Ab = Abf + (size_t)b * T_ * T_;

    const int tid  = threadIdx.x;
    const int lane = tid & 31;
    const int wave = tid >> 5;
    const int wm   = wave >> 1;                  // 0..3 -> m offset wm*32
    const int wn   = wave & 1;                   // 0..1 -> t offset wn*144
    const int lrow = lane & 15;
    const bool hi  = lane >= 16;

    // Issue async copies of the 32(s) x 288(t) A-tile for K-step k0 into sB[buf].
    // 288 rows x 64B = 1152 lane-B128 transfers.
    auto issue_b = [&](int k0, int buf) {
#pragma unroll
        for (int i = 0; i < 5; ++i) {
            int g = tid + i * 256;
            if (g < T_ * 4) {
                int t    = g >> 2;
                int part = g & 3;
                unsigned long long gp = (unsigned long long)(uintptr_t)
                    (Ab + (size_t)t * T_ + k0 + part * 8);
                unsigned lds = (unsigned)(unsigned long long)(uintptr_t)
                    &sB[buf][t * BPITCH + part * 8];
                asm volatile("global_load_async_to_lds_b128 %0, %1, off"
                             :: "v"(lds), "v"(gp) : "memory");
            }
        }
    };

    v8f acc[2][9];
#pragma unroll
    for (int i = 0; i < 2; ++i)
#pragma unroll
        for (int j = 0; j < 9; ++j)
#pragma unroll
            for (int e = 0; e < 8; ++e) acc[i][j][e] = 0.f;

    issue_b(0, 0);                               // prologue: tile 0 in flight

    for (int kk = 0; kk < T_ / BK; ++kk) {
        const int k0  = kk * BK;
        const int cur = kk & 1;

        // tile kk's async writes (this wave's share) are complete
        asm volatile("s_wait_asynccnt 0x0" ::: "memory");
        __syncthreads();   // all waves done writing tile kk + done reading kk-1

        // prefetch next A-tile into the other buffer, overlapped with compute
        if (kk + 1 < T_ / BK) issue_b(k0 + BK, cur ^ 1);

        // ---- stage x tile [BM x BK] fp32 -> bf16 into LDS ----
#pragma unroll
        for (int i = 0; i < 4; ++i) {
            int f4i = tid + i * 256;             // 1024 float4 total
            int row = f4i >> 3;
            int col = (f4i & 7) * 4;
            int n   = m0 + row;
            float4 v = make_float4(0.f, 0.f, 0.f, 0.f);
            if (n < N_) v = *(const float4*)(xbc + (size_t)n * T_ + k0 + col);
            ushort4 h;
            h.x = f2bf(v.x); h.y = f2bf(v.y); h.z = f2bf(v.z); h.w = f2bf(v.w);
            *(ushort4*)&sX[row * XPITCH + col] = h;
        }
        __syncthreads();

        // ---- A fragments (16-bit A layout: lanes<16 K{0..7,16..23}; lanes>=16 K{8..15,24..31})
        BfFrag afrag[2];
#pragma unroll
        for (int mf = 0; mf < 2; ++mf) {
            int row = wm * 32 + mf * 16 + lrow;
            int cb  = hi ? 8 : 0;
            afrag[mf].h[0] = *(const ushort8*)&sX[row * XPITCH + cb];
            afrag[mf].h[1] = *(const ushort8*)&sX[row * XPITCH + cb + 16];
        }

        // ---- B fragments from LDS A-tile (lane = t column; lanes<16 K{0..7,16..23}; lanes>=16 K{8..15,24..31}) ----
#pragma unroll
        for (int nf = 0; nf < 9; ++nf) {
            int tcol = wn * 144 + nf * 16 + lrow;          // N column = t
            const unsigned short* bp = &sB[cur][tcol * BPITCH + (hi ? 8 : 0)];
            BfFrag bfrag;
            bfrag.h[0] = *(const ushort8*)(bp);
            bfrag.h[1] = *(const ushort8*)(bp + 16);
#pragma unroll
            for (int mf = 0; mf < 2; ++mf) {
                acc[mf][nf] = __builtin_amdgcn_wmma_f32_16x16x32_bf16(
                    false, afrag[mf].bf, false, bfrag.bf,
                    (short)0, acc[mf][nf], false, false);
            }
        }
    }

    // ---- store D (f32 16x16 layout: VGPR v -> M = v + 8*hi, N = lane&15) ----
#pragma unroll
    for (int mf = 0; mf < 2; ++mf) {
#pragma unroll
        for (int nf = 0; nf < 9; ++nf) {
            int tcol = wn * 144 + nf * 16 + lrow;
#pragma unroll
            for (int v = 0; v < 8; ++v) {
                int n = m0 + wm * 32 + mf * 16 + v + (hi ? 8 : 0);
                if (n < N_) obc[(size_t)n * T_ + tcol] = acc[mf][nf][v];
            }
        }
    }
}

// ---------------------------------------------------------------------------
// Launch
// ---------------------------------------------------------------------------
extern "C" void kernel_launch(void* const* d_in, const int* in_sizes, int n_in,
                              void* d_out, int out_size, void* d_ws, size_t ws_size,
                              hipStream_t stream) {
    const float* x     = (const float*)d_in[0];   // [B,C,N,T]
    const float* W1    = (const float*)d_in[1];   // [R,N]
    const float* W2    = (const float*)d_in[2];   // [R,N]
    const float* alpha = (const float*)d_in[3];   // [C]
    float* out = (float*)d_out;

    // workspace layout (bytes), all 256-aligned; total ~8.7 MB
    char* ws = (char*)d_ws;
    const size_t kbuf_bytes = (size_t)B_ * N_ * T_ * sizeof(float);     // 5,658,624
    const size_t kw_bytes   = (size_t)B_ * T_ * R_ * sizeof(float);     //   184,320
    float*          Kbuf = (float*)(ws);
    float*          KW1  = (float*)(ws + kbuf_bytes);
    float*          QW2  = (float*)(ws + kbuf_bytes + kw_bytes);
    unsigned short* Abf  = (unsigned short*)(ws + kbuf_bytes + 2 * kw_bytes);

    // 1) channel reduction
    {
        int total = B_ * N_ * T_;
        k_reduce_c<<<(total + 255) / 256, 256, 0, stream>>>(x, alpha, Kbuf);
    }
    // 2) low-rank projections
    k_project<<<B_, 288, 0, stream>>>(Kbuf, W1, W2, KW1, QW2);
    // 3) scores + softmax -> bf16 attention weights
    k_scores_softmax<<<B_ * T_, 288, 0, stream>>>(KW1, QW2, Abf);
    // 4) big WMMA GEMM: out = x @ A^T over the time axis
    {
        dim3 grid((N_ + BM - 1) / BM, C_, B_);   // (3, 64, 16)
        k_av<<<grid, 256, 0, stream>>>(x, Abf, out);
    }
}
